// HISO_73856257622164
// MI455X (gfx1250) — compile-verified, matrix operand order checked by reading, not run on hardware
//
#include <hip/hip_runtime.h>
#include <hip/hip_fp16.h>

typedef _Float16 f16;
typedef __attribute__((ext_vector_type(16))) _Float16 v16h;
typedef __attribute__((ext_vector_type(8)))  _Float16 v8h;
typedef __attribute__((ext_vector_type(8)))  float    v8f;

#define B_  512
#define L_  512
#define DP  128      // channel dim padded 100 -> 128
#define V_  23757
#define P_  57

static __device__ __forceinline__ float sigmoidf_(float x) {
    return 1.0f / (1.0f + __expf(-x));
}
static __device__ __forceinline__ float tanhfast_(float x) {
    x = fminf(15.0f, fmaxf(-15.0f, x));
    float e = __expf(-2.0f * x);
    return (1.0f - e) / (1.0f + e);
}

// ---------------- prep: cast embedding tables to f16, pad channels to 128 ---
__global__ void k_cast_emb(const float* __restrict__ src, f16* __restrict__ dst, int cols) {
    int r = blockIdx.x;
    int c = threadIdx.x;                 // blockDim.x == 128
    float v = (c < cols) ? src[r * cols + c] : 0.0f;
    dst[r * DP + c] = (f16)v;
}

// ---------------- prep: combined conv+wih0 weight for wd path, swizzled -----
// Bswz layout: [(kc*2 + nt)*32 + lane] * 16 f16 elements == exactly the v16h
// fragment each lane feeds to v_wmma_f32_16x16x32_f16 as the B operand.
// k = kc*32 + K_local,  off = k>>7 (tap offset idx 0..3), c = k&127 (channel)
__global__ void k_pack_wdB(const float* __restrict__ wih0,
                           const float* __restrict__ w2, const float* __restrict__ w3,
                           const float* __restrict__ w4, f16* __restrict__ Bswz) {
    int id = blockIdx.x * 256 + threadIdx.x;        // 16*2*32*16 = 16384 total
    if (id >= 16384) return;
    int e = id & 15, lane = (id >> 4) & 31, nt = (id >> 9) & 1, kc = id >> 10;
    int n = nt * 16 + (lane & 15);
    int dir = n >> 4, g = n & 15;
    int half = lane >> 4, v = e >> 1;
    int kl = (v < 4 ? 2 * v : 16 + 2 * (v - 4)) + (e & 1) + 8 * half;
    int k = kc * 32 + kl;
    int off = k >> 7, c = k & 127;
    float acc = 0.0f;
    if (g < 9 && c < 100) {
        const float* wih = wih0 + (dir * 9 + g) * 144;
        for (int j = 0; j < 144; ++j) {
            float wc = 0.0f;
            if (j < 48) {                                   // conv2: taps at off 1,2
                if (off == 1)      wc = w2[(j * 100 + c) * 2 + 0];
                else if (off == 2) wc = w2[(j * 100 + c) * 2 + 1];
            } else if (j < 96) {                            // conv3: taps at off 1,2,3
                int j3 = j - 48;
                if (off >= 1)      wc = w3[(j3 * 100 + c) * 3 + (off - 1)];
            } else {                                        // conv4: taps at off 0..3
                int j4 = j - 96;
                wc = w4[(j4 * 100 + c) * 4 + off];
            }
            acc += wih[j] * wc;
        }
    }
    Bswz[id] = (f16)acc;
}

// combined bias: bg[dir*16+g] = wih0[dir,g,:] . conv_bias + bih0[dir,g]
__global__ void k_pack_wd_bias(const float* __restrict__ wih0, const float* __restrict__ bih0,
                               const float* __restrict__ b2, const float* __restrict__ b3,
                               const float* __restrict__ b4, float* __restrict__ bg) {
    int n = threadIdx.x;
    if (n >= 32) return;
    int dir = n >> 4, g = n & 15;
    float acc = 0.0f;
    if (g < 9) {
        acc = bih0[dir * 9 + g];
        const float* wih = wih0 + (dir * 9 + g) * 144;
        for (int j = 0; j < 144; ++j) {
            float bj = (j < 48) ? b2[j] : (j < 96) ? b3[j - 48] : b4[j - 96];
            acc += wih[j] * bj;
        }
    }
    bg[n] = acc;
}

// swizzled B for pos path: Wg[k=chan][n=dir*16+g] = pos_wih0[dir,g,chan]
__global__ void k_pack_posB(const float* __restrict__ wih0, f16* __restrict__ Bswz) {
    int id = blockIdx.x * 256 + threadIdx.x;        // 4*2*32*16 = 4096 total
    if (id >= 4096) return;
    int e = id & 15, lane = (id >> 4) & 31, nt = (id >> 9) & 1, kc = id >> 10;
    int n = nt * 16 + (lane & 15);
    int dir = n >> 4, g = n & 15;
    int half = lane >> 4, v = e >> 1;
    int kl = (v < 4 ? 2 * v : 16 + 2 * (v - 4)) + (e & 1) + 8 * half;
    int c = kc * 32 + kl;
    float acc = 0.0f;
    if (g < 9 && c < 100) acc = wih0[(dir * 9 + g) * 100 + c];
    Bswz[id] = (f16)acc;
}

// ---------------- WMMA GEMM: wd gx = gather(conv-patch) @ Wg + bg -----------
// block = (t-block of 128, batch b); 8 waves, each owns one 16-row M tile.
__global__ __launch_bounds__(256)
void k_gemm_wd(const int* __restrict__ wd, const f16* __restrict__ embB,
               const f16* __restrict__ Bswz, const float* __restrict__ bg,
               float* __restrict__ gx_f, float* __restrict__ gx_b) {
    __shared__ __align__(16) f16 sA[132 * DP];      // rows t0-2 .. t0+129
    int b = blockIdx.y, t0 = blockIdx.x * 128, tid = threadIdx.x;
    for (int u = tid; u < 132 * 16; u += 256) {     // 16B chunks
        int row = u >> 4, ch = u & 15;
        int tg = t0 - 2 + row;
        uint4 val = {0u, 0u, 0u, 0u};
        if (tg >= 0 && tg < L_) {
            int idx = wd[b * L_ + tg];
            val = *reinterpret_cast<const uint4*>(embB + idx * DP + ch * 8);
        }
        *reinterpret_cast<uint4*>(&sA[row * DP + ch * 8]) = val;
    }
    __syncthreads();

    int wave = tid >> 5, lane = tid & 31;
    int mrow = lane & 15, half = lane >> 4;
    v8f acc0 = {}; v8f acc1 = {};
    for (int kc = 0; kc < 16; ++kc) {
        int off = kc >> 2;
        int kin = (kc & 3) * 32;
        int rowidx = wave * 16 + mrow + off;
        const f16* ap = &sA[rowidx * DP + kin + 8 * half];
        v8h alo = *reinterpret_cast<const v8h*>(ap);
        v8h ahi = *reinterpret_cast<const v8h*>(ap + 16);
        v16h afrag = __builtin_shufflevector(alo, ahi,
                        0,1,2,3,4,5,6,7,8,9,10,11,12,13,14,15);
        const v16h* bp = reinterpret_cast<const v16h*>(Bswz) + (kc * 2) * 32 + lane;
        v16h bf0 = bp[0];
        v16h bf1 = bp[32];
        acc0 = __builtin_amdgcn_wmma_f32_16x16x32_f16(false, afrag, false, bf0,
                                                      (short)0, acc0, false, false);
        acc1 = __builtin_amdgcn_wmma_f32_16x16x32_f16(false, afrag, false, bf1,
                                                      (short)0, acc1, false, false);
    }
    int g = lane & 15;
    if (g < 9) {
        float bia0 = bg[g], bia1 = bg[16 + g];
        for (int r = 0; r < 8; ++r) {
            int t = t0 + wave * 16 + r + 8 * half;
            gx_f[((size_t)t * 9 + g) * B_ + b] = acc0[r] + bia0;
            gx_b[((size_t)t * 9 + g) * B_ + b] = acc1[r] + bia1;
        }
    }
}

// ---------------- WMMA GEMM: pos gx = pos_emb[pos] @ wih0.T + bih0 ----------
__global__ __launch_bounds__(256)
void k_gemm_pos(const int* __restrict__ pos, const f16* __restrict__ embB,
                const f16* __restrict__ Bswz, const float* __restrict__ bih0,
                float* __restrict__ gx_f, float* __restrict__ gx_b) {
    __shared__ __align__(16) f16 sE[P_ * DP];       // whole 57-row table in LDS
    __shared__ int sidx[128];
    int b = blockIdx.y, t0 = blockIdx.x * 128, tid = threadIdx.x;
    for (int u = tid; u < P_ * 16; u += 256) {
        int row = u >> 4, ch = u & 15;
        *reinterpret_cast<uint4*>(&sE[row * DP + ch * 8]) =
            *reinterpret_cast<const uint4*>(embB + row * DP + ch * 8);
    }
    if (tid < 128) sidx[tid] = pos[b * L_ + t0 + tid];
    __syncthreads();

    int wave = tid >> 5, lane = tid & 31;
    int mrow = lane & 15, half = lane >> 4;
    int idx = sidx[wave * 16 + mrow];
    const f16* arow = &sE[idx * DP];
    v8f acc0 = {}; v8f acc1 = {};
    for (int kc = 0; kc < 4; ++kc) {
        const f16* ap = arow + kc * 32 + 8 * half;
        v8h alo = *reinterpret_cast<const v8h*>(ap);
        v8h ahi = *reinterpret_cast<const v8h*>(ap + 16);
        v16h afrag = __builtin_shufflevector(alo, ahi,
                        0,1,2,3,4,5,6,7,8,9,10,11,12,13,14,15);
        const v16h* bp = reinterpret_cast<const v16h*>(Bswz) + (kc * 2) * 32 + lane;
        v16h bf0 = bp[0];
        v16h bf1 = bp[32];
        acc0 = __builtin_amdgcn_wmma_f32_16x16x32_f16(false, afrag, false, bf0,
                                                      (short)0, acc0, false, false);
        acc1 = __builtin_amdgcn_wmma_f32_16x16x32_f16(false, afrag, false, bf1,
                                                      (short)0, acc1, false, false);
    }
    int g = lane & 15;
    if (g < 9) {
        float bi0 = bih0[g], bi1 = bih0[9 + g];
        for (int r = 0; r < 8; ++r) {
            int t = t0 + wave * 16 + r + 8 * half;
            gx_f[((size_t)t * 9 + g) * B_ + b] = acc0[r] + bi0;
            gx_b[((size_t)t * 9 + g) * B_ + b] = acc1[r] + bi1;
        }
    }
}

// ---------------- GRU layer0 scans: 4 direction-jobs via blockIdx.y ---------
__global__ __launch_bounds__(256)
void k_scan0(const float* __restrict__ gxwf, const float* __restrict__ gxwb,
             const float* __restrict__ gxpf, const float* __restrict__ gxpb,
             const float* __restrict__ wd_whh0, const float* __restrict__ wd_bhh0,
             const float* __restrict__ pos_whh0, const float* __restrict__ pos_bhh0,
             float* __restrict__ wd0, float* __restrict__ pos0) {
    int b = blockIdx.x * 256 + threadIdx.x;
    int job = blockIdx.y;
    const float *gx, *whh, *bhh; float* out; int rev, col;
    switch (job) {
        case 0:  gx = gxwf; whh = wd_whh0;       bhh = wd_bhh0;     out = wd0;  rev = 0; col = 0; break;
        case 1:  gx = gxwb; whh = wd_whh0 + 27;  bhh = wd_bhh0 + 9; out = wd0;  rev = 1; col = 3; break;
        case 2:  gx = gxpf; whh = pos_whh0;      bhh = pos_bhh0;    out = pos0; rev = 0; col = 0; break;
        default: gx = gxpb; whh = pos_whh0 + 27; bhh = pos_bhh0 + 9;out = pos0; rev = 1; col = 3; break;
    }
    float W[9][3], Bh[9];
    for (int g = 0; g < 9; ++g) {
        Bh[g] = bhh[g];
        for (int k = 0; k < 3; ++k) W[g][k] = whh[g * 3 + k];
    }
    float h0 = 0.f, h1 = 0.f, h2 = 0.f;
    for (int s = 0; s < L_; ++s) {
        int t = rev ? (L_ - 1 - s) : s;
        const float* gp = gx + (size_t)t * 9 * B_ + b;
        __builtin_prefetch(gp + (rev ? -(9 * B_) : (9 * B_)), 0, 3);
        float x0 = gp[0],      x1 = gp[B_],     x2 = gp[2 * B_];
        float x3 = gp[3 * B_], x4 = gp[4 * B_], x5 = gp[5 * B_];
        float x6 = gp[6 * B_], x7 = gp[7 * B_], x8 = gp[8 * B_];
        float gh[9];
#pragma unroll
        for (int g = 0; g < 9; ++g)
            gh[g] = Bh[g] + W[g][0] * h0 + W[g][1] * h1 + W[g][2] * h2;
        float r0 = sigmoidf_(x0 + gh[0]), r1 = sigmoidf_(x1 + gh[1]), r2 = sigmoidf_(x2 + gh[2]);
        float z0 = sigmoidf_(x3 + gh[3]), z1 = sigmoidf_(x4 + gh[4]), z2 = sigmoidf_(x5 + gh[5]);
        float n0 = tanhfast_(x6 + r0 * gh[6]);
        float n1 = tanhfast_(x7 + r1 * gh[7]);
        float n2 = tanhfast_(x8 + r2 * gh[8]);
        h0 = (1.f - z0) * n0 + z0 * h0;
        h1 = (1.f - z1) * n1 + z1 * h1;
        h2 = (1.f - z2) * n2 + z2 * h2;
        float* op = out + ((size_t)t * 6 + col) * B_ + b;
        op[0] = h0; op[B_] = h1; op[2 * B_] = h2;
    }
}

// ---------------- GRU layer1 scans (inline gx from 6-dim input) -------------
__global__ __launch_bounds__(256)
void k_scan1(const float* __restrict__ wd0, const float* __restrict__ pos0,
             const float* __restrict__ wd_wih1,  const float* __restrict__ wd_whh1,
             const float* __restrict__ wd_bih1,  const float* __restrict__ wd_bhh1,
             const float* __restrict__ pos_wih1, const float* __restrict__ pos_whh1,
             const float* __restrict__ pos_bih1, const float* __restrict__ pos_bhh1,
             float* __restrict__ wd1, float* __restrict__ pos_last) {
    int b = blockIdx.x * 256 + threadIdx.x;
    int job = blockIdx.y;
    const float *in, *wih, *whh, *bih, *bhh; int rev, col, nsteps, seq;
    switch (job) {
        case 0:  in = wd0;  wih = wd_wih1;       whh = wd_whh1;       bih = wd_bih1;      bhh = wd_bhh1;      rev = 0; col = 0; nsteps = L_; seq = 1; break;
        case 1:  in = wd0;  wih = wd_wih1 + 54;  whh = wd_whh1 + 27;  bih = wd_bih1 + 9;  bhh = wd_bhh1 + 9;  rev = 1; col = 3; nsteps = L_; seq = 1; break;
        case 2:  in = pos0; wih = pos_wih1;      whh = pos_whh1;      bih = pos_bih1;     bhh = pos_bhh1;     rev = 0; col = 0; nsteps = L_; seq = 0; break;
        default: in = pos0; wih = pos_wih1 + 54; whh = pos_whh1 + 27; bih = pos_bih1 + 9; bhh = pos_bhh1 + 9; rev = 1; col = 3; nsteps = 1;  seq = 0; break;
    }
    float Wi[9][6], W[9][3], Bi[9], Bh[9];
    for (int g = 0; g < 9; ++g) {
        Bi[g] = bih[g]; Bh[g] = bhh[g];
        for (int c = 0; c < 6; ++c) Wi[g][c] = wih[g * 6 + c];
        for (int k = 0; k < 3; ++k) W[g][k] = whh[g * 3 + k];
    }
    float h0 = 0.f, h1 = 0.f, h2 = 0.f;
    for (int s = 0; s < nsteps; ++s) {
        int t = rev ? (L_ - 1 - s) : s;
        const float* ip = in + (size_t)t * 6 * B_ + b;
        float x[6];
#pragma unroll
        for (int c = 0; c < 6; ++c) x[c] = ip[c * B_];
        float gx[9];
#pragma unroll
        for (int g = 0; g < 9; ++g) {
            float a = Bi[g];
#pragma unroll
            for (int c = 0; c < 6; ++c) a += Wi[g][c] * x[c];
            gx[g] = a;
        }
        float gh[9];
#pragma unroll
        for (int g = 0; g < 9; ++g)
            gh[g] = Bh[g] + W[g][0] * h0 + W[g][1] * h1 + W[g][2] * h2;
        float r0 = sigmoidf_(gx[0] + gh[0]), r1 = sigmoidf_(gx[1] + gh[1]), r2 = sigmoidf_(gx[2] + gh[2]);
        float z0 = sigmoidf_(gx[3] + gh[3]), z1 = sigmoidf_(gx[4] + gh[4]), z2 = sigmoidf_(gx[5] + gh[5]);
        float n0 = tanhfast_(gx[6] + r0 * gh[6]);
        float n1 = tanhfast_(gx[7] + r1 * gh[7]);
        float n2 = tanhfast_(gx[8] + r2 * gh[8]);
        h0 = (1.f - z0) * n0 + z0 * h0;
        h1 = (1.f - z1) * n1 + z1 * h1;
        h2 = (1.f - z2) * n2 + z2 * h2;
        if (seq) {
            float* op = wd1 + ((size_t)t * 6 + col) * B_ + b;
            op[0] = h0; op[B_] = h1; op[2 * B_] = h2;
        }
    }
    if (!seq) {
        pos_last[b * 6 + col + 0] = h0;
        pos_last[b * 6 + col + 1] = h1;
        pos_last[b * 6 + col + 2] = h2;
    }
}

// ---------------- attention pooling over L, one block per batch row ---------
__global__ __launch_bounds__(256)
void k_atten(const float* __restrict__ wd1, const float* __restrict__ squish_w,
             const float* __restrict__ atten_proj, float* __restrict__ wd_atten) {
    __shared__ float satt[L_];
    __shared__ float sred[256];
    int b = blockIdx.x, tid = threadIdx.x;
    float sw[36], ap[6];
    for (int i = 0; i < 36; ++i) sw[i] = squish_w[i];
    for (int j = 0; j < 6; ++j) ap[j] = atten_proj[j];
    for (int t = tid; t < L_; t += 256) {
        const float* wp = wd1 + (size_t)t * 6 * B_ + b;
        float wo[6];
        for (int c = 0; c < 6; ++c) wo[c] = wp[c * B_];
        float a = 0.f;
        for (int j = 0; j < 6; ++j) {
            float s = 0.f;
            for (int c = 0; c < 6; ++c) s += wo[c] * sw[c * 6 + j];
            a += tanhfast_(s) * ap[j];
        }
        satt[t] = a;
    }
    __syncthreads();
    sred[tid] = fmaxf(satt[tid], satt[tid + 256]);
    __syncthreads();
    for (int st = 128; st > 0; st >>= 1) {
        if (tid < st) sred[tid] = fmaxf(sred[tid], sred[tid + st]);
        __syncthreads();
    }
    float m = sred[0];
    __syncthreads();
    float psum = 0.f;
    for (int t = tid; t < L_; t += 256) {
        float e = __expf(satt[t] - m);
        satt[t] = e;
        psum += e;
    }
    sred[tid] = psum;
    __syncthreads();
    for (int st = 128; st > 0; st >>= 1) {
        if (tid < st) sred[tid] += sred[tid + st];
        __syncthreads();
    }
    float inv = 1.f / sred[0];
    float p[6] = {0.f, 0.f, 0.f, 0.f, 0.f, 0.f};
    for (int t = tid; t < L_; t += 256) {
        float w = satt[t];
        const float* wp = wd1 + (size_t)t * 6 * B_ + b;
        for (int c = 0; c < 6; ++c) p[c] += w * wp[c * B_];
    }
    for (int c = 0; c < 6; ++c) {
        __syncthreads();
        sred[tid] = p[c];
        __syncthreads();
        for (int st = 128; st > 0; st >>= 1) {
            if (tid < st) sred[tid] += sred[tid + st];
            __syncthreads();
        }
        if (tid == 0) wd_atten[b * 6 + c] = sred[0] * inv;
    }
}

// ---------------- final head: bn+relu, aux softmax, fc1->fc2, softmax -------
__global__ __launch_bounds__(256)
void k_head(const float* __restrict__ pos_last, const float* __restrict__ wd_atten,
            const float* __restrict__ posfc_gamma, const float* __restrict__ posfc_beta,
            const float* __restrict__ posfc_w, const float* __restrict__ posfc_b,
            const float* __restrict__ fc1_w, const float* __restrict__ fc1_b,
            const float* __restrict__ fc_gamma, const float* __restrict__ fc_beta,
            const float* __restrict__ fc2_w, const float* __restrict__ fc2_b,
            float* __restrict__ out) {
    int b = blockIdx.x * 256 + threadIdx.x;
    const float inv = rsqrtf(1.0f + 1e-5f);
    float ph[6];
    for (int c = 0; c < 6; ++c) {
        float v = pos_last[b * 6 + c] * inv * posfc_gamma[c] + posfc_beta[c];
        ph[c] = fmaxf(v, 0.f);
    }
    float a[3]; float amax = -1e30f;
    for (int o = 0; o < 3; ++o) {
        float s = posfc_b[o];
        for (int c = 0; c < 6; ++c) s += posfc_w[o * 6 + c] * ph[c];
        a[o] = s; amax = fmaxf(amax, s);
    }
    float asum = 0.f;
    for (int o = 0; o < 3; ++o) { a[o] = __expf(a[o] - amax); asum += a[o]; }
    for (int o = 0; o < 3; ++o) a[o] /= asum;
    float comb[9];
    for (int c = 0; c < 6; ++c) comb[c] = wd_atten[b * 6 + c];
    for (int o = 0; o < 3; ++o) comb[6 + o] = a[o];
    float lg[6];
    for (int o = 0; o < 6; ++o) lg[o] = fc2_b[o];
    for (int i = 0; i < 128; ++i) {
        float s = fc1_b[i];
        for (int j = 0; j < 9; ++j) s += fc1_w[i * 9 + j] * comb[j];
        float hv = fmaxf(s * inv * fc_gamma[i] + fc_beta[i], 0.f);
        for (int o = 0; o < 6; ++o) lg[o] += fc2_w[o * 128 + i] * hv;
    }
    float lmax = -1e30f;
    for (int o = 0; o < 6; ++o) lmax = fmaxf(lmax, lg[o]);
    float lsum = 0.f;
    for (int o = 0; o < 6; ++o) { lg[o] = __expf(lg[o] - lmax); lsum += lg[o]; }
    for (int o = 0; o < 6; ++o) out[b * 6 + o] = lg[o] / lsum;
    for (int o = 0; o < 3; ++o) out[B_ * 6 + b * 3 + o] = a[o];
}

// ---------------------------------------------------------------------------
extern "C" void kernel_launch(void* const* d_in, const int* in_sizes, int n_in,
                              void* d_out, int out_size, void* d_ws, size_t ws_size,
                              hipStream_t stream) {
    const int*   wd         = (const int*)  d_in[0];
    const int*   pos        = (const int*)  d_in[1];
    const float* wd_emb     = (const float*)d_in[2];
    const float* pos_emb    = (const float*)d_in[3];
    const float* conv_w2    = (const float*)d_in[4];
    const float* conv_b2    = (const float*)d_in[5];
    const float* conv_w3    = (const float*)d_in[6];
    const float* conv_b3    = (const float*)d_in[7];
    const float* conv_w4    = (const float*)d_in[8];
    const float* conv_b4    = (const float*)d_in[9];
    const float* wd_wih0    = (const float*)d_in[10];
    const float* wd_whh0    = (const float*)d_in[11];
    const float* wd_bih0    = (const float*)d_in[12];
    const float* wd_bhh0    = (const float*)d_in[13];
    const float* wd_wih1    = (const float*)d_in[14];
    const float* wd_whh1    = (const float*)d_in[15];
    const float* wd_bih1    = (const float*)d_in[16];
    const float* wd_bhh1    = (const float*)d_in[17];
    const float* pos_wih0   = (const float*)d_in[18];
    const float* pos_whh0   = (const float*)d_in[19];
    const float* pos_bih0   = (const float*)d_in[20];
    const float* pos_bhh0   = (const float*)d_in[21];
    const float* pos_wih1   = (const float*)d_in[22];
    const float* pos_whh1   = (const float*)d_in[23];
    const float* pos_bih1   = (const float*)d_in[24];
    const float* pos_bhh1   = (const float*)d_in[25];
    const float* squish_w   = (const float*)d_in[26];
    const float* atten_proj = (const float*)d_in[27];
    const float* posfc_g    = (const float*)d_in[28];
    const float* posfc_be   = (const float*)d_in[29];
    const float* posfc_w    = (const float*)d_in[30];
    const float* posfc_b    = (const float*)d_in[31];
    const float* fc1_w      = (const float*)d_in[32];
    const float* fc1_b      = (const float*)d_in[33];
    const float* fc_g       = (const float*)d_in[34];
    const float* fc_be      = (const float*)d_in[35];
    const float* fc2_w      = (const float*)d_in[36];
    const float* fc2_b      = (const float*)d_in[37];
    float* out = (float*)d_out;

    char* ws = (char*)d_ws;
    size_t o = 0;
    auto alloc = [&](size_t bytes) -> char* {
        char* p = ws + o;
        o += (bytes + 255) & ~(size_t)255;
        return p;
    };
    f16*   wdEmbB   = (f16*)  alloc((size_t)V_ * DP * sizeof(f16));
    f16*   posEmbB  = (f16*)  alloc((size_t)P_ * DP * sizeof(f16));
    f16*   BswzWd   = (f16*)  alloc(16384 * sizeof(f16));
    float* bgWd     = (float*)alloc(32 * sizeof(float));
    f16*   BswzPos  = (f16*)  alloc(4096 * sizeof(f16));
    float* gxwf     = (float*)alloc((size_t)L_ * 9 * B_ * sizeof(float));
    float* gxwb     = (float*)alloc((size_t)L_ * 9 * B_ * sizeof(float));
    float* gxpf     = (float*)alloc((size_t)L_ * 9 * B_ * sizeof(float));
    float* gxpb     = (float*)alloc((size_t)L_ * 9 * B_ * sizeof(float));
    float* wd0      = (float*)alloc((size_t)L_ * 6 * B_ * sizeof(float));
    float* pos0     = (float*)alloc((size_t)L_ * 6 * B_ * sizeof(float));
    float* wd1      = (float*)alloc((size_t)L_ * 6 * B_ * sizeof(float));
    float* posLast  = (float*)alloc((size_t)B_ * 6 * sizeof(float));
    float* wdAtten  = (float*)alloc((size_t)B_ * 6 * sizeof(float));

    k_cast_emb<<<V_, 128, 0, stream>>>(wd_emb, wdEmbB, 100);
    k_cast_emb<<<P_, 128, 0, stream>>>(pos_emb, posEmbB, 100);
    k_pack_wdB<<<64, 256, 0, stream>>>(wd_wih0, conv_w2, conv_w3, conv_w4, BswzWd);
    k_pack_wd_bias<<<1, 32, 0, stream>>>(wd_wih0, wd_bih0, conv_b2, conv_b3, conv_b4, bgWd);
    k_pack_posB<<<16, 256, 0, stream>>>(pos_wih0, BswzPos);

    k_gemm_wd<<<dim3(4, B_), 256, 0, stream>>>(wd, wdEmbB, BswzWd, bgWd, gxwf, gxwb);
    k_gemm_pos<<<dim3(4, B_), 256, 0, stream>>>(pos, posEmbB, BswzPos, pos_bih0, gxpf, gxpb);

    k_scan0<<<dim3(2, 4), 256, 0, stream>>>(gxwf, gxwb, gxpf, gxpb,
                                            wd_whh0, wd_bhh0, pos_whh0, pos_bhh0,
                                            wd0, pos0);
    k_scan1<<<dim3(2, 4), 256, 0, stream>>>(wd0, pos0,
                                            wd_wih1, wd_whh1, wd_bih1, wd_bhh1,
                                            pos_wih1, pos_whh1, pos_bih1, pos_bhh1,
                                            wd1, posLast);

    k_atten<<<B_, 256, 0, stream>>>(wd1, squish_w, atten_proj, wdAtten);
    k_head<<<2, 256, 0, stream>>>(posLast, wdAtten, posfc_g, posfc_be, posfc_w, posfc_b,
                                  fc1_w, fc1_b, fc_g, fc_be, fc2_w, fc2_b, out);
}
